// IREstimator_84284438217392
// MI455X (gfx1250) — compile-verified, matrix operand order checked by reading, not run on hardware
//
#include <hip/hip_runtime.h>
#include <math.h>

typedef __attribute__((ext_vector_type(16))) _Float16 v16h;
typedef __attribute__((ext_vector_type(8)))  float    v8f;

#define B_TOT  16384
#define D_IN   80
#define NT     10
#define DMODEL 128
#define DHID   512
#define NBLK   3
#define CTX    2
#define BT     16           // batch elements per workgroup (probe: 108KB LDS)
#define ROWS   (BT * NT)    // 160 rows of (b, n)
#define MT     (ROWS / 16)  // 10 m-tiles

// ---- shared memory layout (bytes) ----
// steady state:
//   xs   f16[160*128] @ 0       (40960)  activation tile, rows (b*10+n) x d
//   as   f16[160*128] @ 40960   (40960)  LN output (A matrix of GEMM1)
//   hs   f16[160*32]  @ 81920   (10240)  GELU'd hidden chunk (A of GEMM2)
//   w1s  f16[32*128]  @ 92160   (8192)   pw1 chunk (B of GEMM1, N-major)
//   w2s  f16[128*32]  @ 100352  (8192)   pw2 chunk (B of GEMM2, N-major)
// input-projection overlays (as/hs/w1s/w2s dead):
//   wpj  f16[128*96]  @ 40960   (24576)
//   melA f16[160*96]  @ 65536   (30720)
// head overlays: pooled f32[16*128] @ 81920, h1 f32[16*64] @ 92160
#define OFF_X     0
#define OFF_A     40960
#define OFF_H     81920
#define OFF_W1    92160
#define OFF_W2    100352
#define OFF_WPROJ 40960
#define OFF_MELA  65536
#define SMEM_BYTES 108544

// Load a 16x32 f16 WMMA operand fragment from an LDS tile.
// ISA 7.12.2: lanes 0-15 -> row = lane, lanes 16-31 -> same rows with K+8;
// VGPR v holds K pair {2v,2v+1} (v<4) / {16+2(v-4),...} (v>=4).
__device__ inline v16h load_frag(const _Float16* __restrict__ tile, int stride, int lane) {
    const _Float16* rp = tile + (lane & 15) * stride + ((lane >> 4) ? 8 : 0);
    v16h r;
#pragma unroll
    for (int v = 0; v < 8; ++v) {
        const int k = 2 * v + ((2 * v) >= 8 ? 8 : 0);
        r[2 * v]     = rp[k];
        r[2 * v + 1] = rp[k + 1];
    }
    return r;
}

// Fast activations on the TRANS pipe (v_rcp_f32 / v_exp_f32 / v_tanh_f32);
// avoids the IEEE div_scale+newton expansion seen in round-1 asm.
__device__ inline float fast_rcp(float x) { return __builtin_amdgcn_rcpf(x); }
__device__ inline float silu_f(float x) { return x * fast_rcp(1.f + __expf(-x)); }
__device__ inline float sigm_f(float x) { return fast_rcp(1.f + __expf(-x)); }
__device__ inline float gelu_f(float x) { return 0.5f * x * (1.f + erff(x * 0.70710678118f)); }
__device__ inline float tanh_fast(float x) {
    float r;
    asm volatile("v_tanh_f32 %0, %1\n\tv_nop" : "=v"(r) : "v"(x));  // TRANS hazard: 1 indep op
    return r;
}

__global__ __launch_bounds__(256) void ir_estimator_fused(
    const float* __restrict__ mel,    const float* __restrict__ state_in,
    const float* __restrict__ w_in,   const float* __restrict__ b_in,
    const float* __restrict__ dw_w,   const float* __restrict__ dw_b,
    const float* __restrict__ ln_g,   const float* __restrict__ ln_b,
    const float* __restrict__ pw1_w,  const float* __restrict__ pw1_b,
    const float* __restrict__ pw2_w,  const float* __restrict__ pw2_b,
    const float* __restrict__ mw1,    const float* __restrict__ mb1,
    const float* __restrict__ mw2,    const float* __restrict__ mb2,
    float* __restrict__ out)
{
    __shared__ __align__(16) unsigned char smem[SMEM_BYTES];
    _Float16* xs  = (_Float16*)(smem + OFF_X);
    _Float16* as  = (_Float16*)(smem + OFF_A);
    _Float16* hs  = (_Float16*)(smem + OFF_H);
    _Float16* w1s = (_Float16*)(smem + OFF_W1);
    _Float16* w2s = (_Float16*)(smem + OFF_W2);

    const int tid  = threadIdx.x;
    const int wave = tid >> 5;
    const int lane = tid & 31;
    const int b0   = blockIdx.x * BT;
    const size_t STATE_OUT = (size_t)B_TOT * 32;

    // ================= input projection: x = silu(W_in @ mel + b) =================
    {
        _Float16* melA = (_Float16*)(smem + OFF_MELA);
        _Float16* wpj  = (_Float16*)(smem + OFF_WPROJ);
        for (int idx = tid; idx < ROWS * 96; idx += 256) {
            int r = idx / 96, c = idx % 96;
            int b = r / NT,  n = r % NT;
            float v = (c < D_IN) ? mel[((size_t)(b0 + b) * D_IN + c) * NT + n] : 0.f;
            melA[idx] = (_Float16)v;
        }
        for (int idx = tid; idx < DMODEL * 96; idx += 256) {
            int d = idx / 96, c = idx % 96;
            float v = (c < D_IN) ? w_in[d * D_IN + c] : 0.f;
            wpj[idx] = (_Float16)v;
        }
        __syncthreads();

        for (int p = wave; p < MT * 8; p += 8) {   // uniform per wave
            const int mtile = p >> 3, ntile = p & 7;
            v8f acc = {};
#pragma unroll
            for (int ks = 0; ks < 3; ++ks) {       // K = 80 padded to 96
                v16h a  = load_frag(melA + mtile * 16 * 96 + ks * 32, 96, lane);
                v16h bw = load_frag(wpj  + ntile * 16 * 96 + ks * 32, 96, lane);
                acc = __builtin_amdgcn_wmma_f32_16x16x32_f16(
                          false, a, false, bw, (short)0, acc, false, false);
            }
            const int nn = lane & 15, hi = lane >> 4;
            const int d = ntile * 16 + nn;
            const float bias = b_in[d];
#pragma unroll
            for (int v = 0; v < 8; ++v) {
                int r = mtile * 16 + v + 8 * hi;
                xs[r * DMODEL + d] = (_Float16)silu_f(acc[v] + bias);
            }
        }
        __syncthreads();
    }

    // ================= 3 residual blocks =================
    for (int blk = 0; blk < NBLK; ++blk) {
        // (a) emit new conv state = x[:, :, 8:10] at block entry
        for (int idx = tid; idx < BT * DMODEL * CTX; idx += 256) {
            int j = idx % CTX;
            int d = (idx / CTX) % DMODEL;
            int b = idx / (CTX * DMODEL);
            out[STATE_OUT + ((size_t)(b0 + b) * DMODEL + d) * (NBLK * CTX) + blk * CTX + j] =
                (float)xs[(b * NT + (NT - CTX) + j) * DMODEL + d];
        }

        // (b) depthwise conv (K=3, causal via state) + LayerNorm -> as (f16)
        {
            float cw0[4], cw1[4], cw2[4], cb[4], cg[4], cbt[4];  // row-invariant, hoisted
#pragma unroll
            for (int q = 0; q < 4; ++q) {
                const int d = lane + 32 * q;
                cw0[q] = dw_w[(blk * DMODEL + d) * 3 + 0];
                cw1[q] = dw_w[(blk * DMODEL + d) * 3 + 1];
                cw2[q] = dw_w[(blk * DMODEL + d) * 3 + 2];
                cb[q]  = dw_b[blk * DMODEL + d];
                cg[q]  = ln_g[blk * DMODEL + d];
                cbt[q] = ln_b[blk * DMODEL + d];
            }
            for (int r = wave; r < ROWS; r += 8) {     // one wave per row
                const int b = r / NT, n = r % NT;
                float yv[4];
                float s1 = 0.f, s2 = 0.f;
#pragma unroll
                for (int q = 0; q < 4; ++q) {
                    const int d = lane + 32 * q;
                    float xv[3];
#pragma unroll
                    for (int k = 0; k < 3; ++k) {
                        const int t = n - 2 + k;
                        xv[k] = (t < 0)
                            ? state_in[((size_t)(b0 + b) * DMODEL + d) * (NBLK * CTX) + blk * CTX + (t + 2)]
                            : (float)xs[(b * NT + t) * DMODEL + d];
                    }
                    float y = cw0[q] * xv[0] + cw1[q] * xv[1] + cw2[q] * xv[2] + cb[q];
                    yv[q] = y;
                    s1 += y; s2 += y * y;
                }
#pragma unroll
                for (int off = 16; off > 0; off >>= 1) {   // wave32 allreduce
                    s1 += __shfl_xor(s1, off, 32);
                    s2 += __shfl_xor(s2, off, 32);
                }
                const float mean = s1 * (1.f / DMODEL);
                const float var  = s2 * (1.f / DMODEL) - mean * mean;
                const float rstd = rsqrtf(var + 1e-5f);
#pragma unroll
                for (int q = 0; q < 4; ++q) {
                    const int d = lane + 32 * q;
                    as[r * DMODEL + d] = (_Float16)((yv[q] - mean) * rstd * cg[q] + cbt[q]);
                }
            }
        }
        __syncthreads();

        // (c) MLP 128 -> 512 -> 128 with residual, chunked over hidden dim.
        // wave w owns output d-tile w (all 10 m-tiles); accumulators carry x + b2.
        v8f oacc[MT];
        {
            const int nn = lane & 15, hi = lane >> 4;
            const int d = wave * 16 + nn;
            const float b2 = pw2_b[blk * DMODEL + d];
#pragma unroll
            for (int m = 0; m < MT; ++m) {
#pragma unroll
                for (int v = 0; v < 8; ++v) {
                    int r = m * 16 + v + 8 * hi;
                    oacc[m][v] = (float)xs[r * DMODEL + d] + b2;
                }
            }
        }

        for (int jc = 0; jc < DHID / 32; ++jc) {
            const int j0 = jc * 32;
            // stage W1 chunk (32 x 128) and W2 chunk (128 x 32), N-major f16
            for (int idx = tid; idx < 32 * DMODEL; idx += 256) {
                int jj = idx / DMODEL, k = idx % DMODEL;
                w1s[idx] = (_Float16)pw1_w[((size_t)blk * DHID + j0 + jj) * DMODEL + k];
            }
            for (int idx = tid; idx < DMODEL * 32; idx += 256) {
                int d = idx / 32, k = idx % 32;
                w2s[idx] = (_Float16)pw2_w[((size_t)blk * DMODEL + d) * DHID + j0 + k];
            }
            // prefetch next chunk's weights into cache while we compute this one
            if (jc + 1 < DHID / 32) {
                const int jn = j0 + 32;
                if (tid < 128) {
                    __builtin_prefetch(pw1_w + ((size_t)blk * DHID + jn) * DMODEL + tid * 32, 0, 0);
                } else {
                    __builtin_prefetch(pw2_w + ((size_t)blk * DMODEL + (tid - 128)) * DHID + jn, 0, 0);
                }
            }
            __syncthreads();

            // GEMM1: (160x128) @ (128x32) -> GELU -> hs
            for (int t = wave; t < MT * 2; t += 8) {
                const int m = t >> 1, nt = t & 1;
                v8f acc = {};
#pragma unroll
                for (int ks = 0; ks < 4; ++ks) {
                    v16h a  = load_frag(as  + m  * 16 * DMODEL + ks * 32, DMODEL, lane);
                    v16h bw = load_frag(w1s + nt * 16 * DMODEL + ks * 32, DMODEL, lane);
                    acc = __builtin_amdgcn_wmma_f32_16x16x32_f16(
                              false, a, false, bw, (short)0, acc, false, false);
                }
                const int nn = lane & 15, hi = lane >> 4;
                const int jj = nt * 16 + nn;
                const float b1 = pw1_b[blk * DHID + j0 + jj];
#pragma unroll
                for (int v = 0; v < 8; ++v) {
                    int r = m * 16 + v + 8 * hi;
                    hs[r * 32 + jj] = (_Float16)gelu_f(acc[v] + b1);
                }
            }
            __syncthreads();

            // GEMM2: (160x32) @ (32x128) accumulated into oacc; B-frag loaded once
            {
                v16h bw = load_frag(w2s + wave * 16 * 32, 32, lane);
#pragma unroll
                for (int m = 0; m < MT; ++m) {
                    v16h a = load_frag(hs + m * 16 * 32, 32, lane);
                    oacc[m] = __builtin_amdgcn_wmma_f32_16x16x32_f16(
                                  false, a, false, bw, (short)0, oacc[m], false, false);
                }
            }
            __syncthreads();
        }

        // write residual-updated x back
        {
            const int nn = lane & 15, hi = lane >> 4;
            const int d = wave * 16 + nn;
#pragma unroll
            for (int m = 0; m < MT; ++m) {
#pragma unroll
                for (int v = 0; v < 8; ++v) {
                    int r = m * 16 + v + 8 * hi;
                    xs[r * DMODEL + d] = (_Float16)oacc[m][v];
                }
            }
        }
        __syncthreads();
    }

    // ================= mean pool + MLP head + param activations =================
    float* pooled = (float*)(smem + OFF_H);    // 16*128 f32
    for (int idx = tid; idx < BT * DMODEL; idx += 256) {
        int b = idx / DMODEL, d = idx % DMODEL;
        float s = 0.f;
#pragma unroll
        for (int n = 0; n < NT; ++n) s += (float)xs[(b * NT + n) * DMODEL + d];
        pooled[idx] = s * (1.f / NT);
    }
    __syncthreads();

    float* h1 = (float*)(smem + OFF_W1);       // 16*64 f32
    for (int idx = tid; idx < BT * 64; idx += 256) {
        int b = idx / 64, j = idx % 64;
        float s = mb1[j];
        for (int k = 0; k < DMODEL; ++k) s += pooled[b * DMODEL + k] * mw1[j * DMODEL + k];
        h1[idx] = silu_f(s);
    }
    __syncthreads();

    for (int idx = tid; idx < BT * 32; idx += 256) {
        int b = idx / 32, o = idx % 32;
        float s = mb2[o];
        for (int k = 0; k < 64; ++k) s += h1[b * 64 + k] * mw2[o * 64 + k];
        const float sg = sigm_f(s);
        const float th = tanh_fast(s);
        float r;
        if      (o < 8)  r = sg * 2.95f + 0.05f;
        else if (o < 16) r = sg * 40.f - 10.f;
        else if (o < 24) r = th * 6.f;
        else if (o < 26) r = sg;
        else if (o < 28) r = th;
        else if (o < 30) r = sg * 0.1f;
        else if (o < 31) r = th;
        else             r = sg;
        out[(size_t)(b0 + b) * 32 + o] = r;
    }
}

extern "C" void kernel_launch(void* const* d_in, const int* in_sizes, int n_in,
                              void* d_out, int out_size, void* d_ws, size_t ws_size,
                              hipStream_t stream) {
    (void)in_sizes; (void)n_in; (void)out_size; (void)d_ws; (void)ws_size;
    const float* mel      = (const float*)d_in[0];
    const float* state_in = (const float*)d_in[1];
    const float* w_in     = (const float*)d_in[2];
    const float* b_in     = (const float*)d_in[3];
    const float* dw_w     = (const float*)d_in[4];
    const float* dw_b     = (const float*)d_in[5];
    const float* ln_g     = (const float*)d_in[6];
    const float* ln_b     = (const float*)d_in[7];
    const float* pw1_w    = (const float*)d_in[8];
    const float* pw1_b    = (const float*)d_in[9];
    const float* pw2_w    = (const float*)d_in[10];
    const float* pw2_b    = (const float*)d_in[11];
    const float* mw1      = (const float*)d_in[12];
    const float* mb1      = (const float*)d_in[13];
    const float* mw2      = (const float*)d_in[14];
    const float* mb2      = (const float*)d_in[15];
    float* out            = (float*)d_out;

    dim3 grid(B_TOT / BT), block(256);
    hipLaunchKernelGGL(ir_estimator_fused, grid, block, 0, stream,
                       mel, state_in, w_in, b_in, dw_w, dw_b, ln_g, ln_b,
                       pw1_w, pw1_b, pw2_w, pw2_b, mw1, mb1, mw2, mb2, out);
}
